// STHAFEncoder_85040352461508
// MI455X (gfx1250) — matrix operand order repaired and checked by reading, not compiled, not run on hardware
//
#include <hip/hip_runtime.h>
#include <hip/hip_bf16.h>
#include <math.h>

// ---------------- constants ----------------
#define TT      8
#define NND     2048
#define EE      32768
#define E2C     (EE + NND)        // edges + self loops = 34816
#define INDIM   64
#define HIDC    128
#define HEADSC  4
#define GOUT    512               // HID*HEADS
#define NTN     (TT * NND)        // 16384
#define KPOOL   256
#define NHTF    4
#define LNEPS   1e-5f

typedef __attribute__((ext_vector_type(16))) __bf16 v16bf;
typedef __attribute__((ext_vector_type(8)))  __bf16 v8bf;
typedef __attribute__((ext_vector_type(4)))  __bf16 v4bf;
typedef __attribute__((ext_vector_type(8)))  float  v8f;

// ---------------- helpers ----------------
__device__ __forceinline__ unsigned fenc(float f) {
    unsigned u = __float_as_uint(f);
    return (u & 0x80000000u) ? ~u : (u | 0x80000000u);
}
__device__ __forceinline__ float fdec(unsigned u) {
    return __uint_as_float((u & 0x80000000u) ? (u & 0x7fffffffu) : ~u);
}
__device__ __forceinline__ float wave_sum(float v) {
    #pragma unroll
    for (int off = 16; off; off >>= 1) v += __shfl_xor(v, off, 32);
    return v;
}
__device__ __forceinline__ float sigmf(float x) { return 1.0f / (1.0f + __expf(-x)); }

// ---------------- f32 -> bf16 row-major convert (A operand staging) ----------------
__global__ void cvt_bf16_kernel(const float* __restrict__ s, __bf16* __restrict__ d, int n4)
{
    int i = blockIdx.x * blockDim.x + threadIdx.x;      // 4 elements per thread
    if (i >= n4) return;
    float4 v = ((const float4*)s)[i];
    v4bf o;
    o[0] = (__bf16)v.x; o[1] = (__bf16)v.y; o[2] = (__bf16)v.z; o[3] = (__bf16)v.w;
    ((v4bf*)d)[i] = o;
}

// ---------------- pack B (f32, opt. transposed) into fragment-major bf16 ----------------
// Fragment layout: for (kblk, ntile): 32 lanes x 16 bf16 contiguous (32B per lane).
// Lane l: column n = ntile*16 + (l&15); half=(l>>4); element 2v(+1) = B(k,n),
// k = kblk*32 + half*16 + 2v(+1).  Contiguous k order, same as the A fragment in wmma_gemm_kernel.
__global__ void pack_b_kernel(const float* __restrict__ B, __bf16* __restrict__ Bp,
                              int Nc, int K, int ldb, int transB)
{
    int idx = blockIdx.x * blockDim.x + threadIdx.x;    // one (fragment,lane) per thread
    int ntiles = Nc >> 4;
    int total = (K >> 5) * ntiles * 32;
    if (idx >= total) return;
    int lane = idx & 31;
    int frag = idx >> 5;
    int ntile = frag % ntiles;
    int kblk  = frag / ntiles;
    int n = ntile * 16 + (lane & 15);
    int half = lane >> 4;
    v16bf o;
    #pragma unroll
    for (int v = 0; v < 8; ++v) {
        int k = kblk * 32 + half * 16 + v * 2;
        float b0 = transB ? B[(size_t)n * ldb + k]     : B[(size_t)k * ldb + n];
        float b1 = transB ? B[(size_t)n * ldb + k + 1] : B[(size_t)(k + 1) * ldb + n];
        o[2 * v]     = (__bf16)b0;
        o[2 * v + 1] = (__bf16)b1;
    }
    *(v16bf*)(Bp + (size_t)idx * 16) = o;
}

// ---------------- WMMA GEMM: C = act(Abf @ Bp + bias) ----------------
// Abf: [M,K] bf16 row-major. Bp: fragment-major packed bf16 (see pack_b_kernel).
// One wave32 owns a 16x64 strip. Software-pipelined: next K-step's A + 4 B
// fragments are fetched into fresh registers while the current step's four
// independent v_wmma_f32_16x16x32_bf16 execute.
template <int ACT>
__global__ void wmma_gemm_kernel(const __bf16* __restrict__ A, const __bf16* __restrict__ Bp,
                                 const float* __restrict__ bias, float* __restrict__ C,
                                 int M, int Nc, int K)
{
    int wave = (blockIdx.x * blockDim.x + threadIdx.x) >> 5;
    int lane = threadIdx.x & 31;
    int stripsN = Nc >> 6;                       // strips of 64 columns
    int strips  = (M >> 4) * stripsN;
    if (wave >= strips) return;                  // wave-uniform: EXEC all-ones for WMMA
    int tm = wave / stripsN, tn = wave % stripsN;
    int row0 = tm << 4, col0 = tn << 6;
    int ntiles = Nc >> 4;
    int tile0  = tn << 2;                        // first of 4 N-tiles
    int half = lane >> 4;                        // K half-wave selector
    int l15  = lane & 15;
    int nk = K >> 5;

    const __bf16* Arow   = A + (size_t)(row0 + l15) * K + 16 * half;
    const __bf16* fbBase = Bp + ((size_t)tile0 * 32 + lane) * 16;
    const size_t  fstep  = (size_t)ntiles * 32 * 16;    // elements per kblk

    // prologue: fragments for kblk 0
    v8bf  alo = *(const v8bf*)(Arow);
    v8bf  ahi = *(const v8bf*)(Arow + 8);
    v16bf b0  = *(const v16bf*)(fbBase);
    v16bf b1  = *(const v16bf*)(fbBase + 512);
    v16bf b2  = *(const v16bf*)(fbBase + 1024);
    v16bf b3  = *(const v16bf*)(fbBase + 1536);

    v8f c0 = {}, c1 = {}, c2 = {}, c3 = {};
    for (int kblk = 0; kblk < nk; ++kblk) {
        // issue next iteration's loads first (overlap with WMMAs below)
        v8bf alo_n = alo, ahi_n = ahi;
        v16bf b0n = b0, b1n = b1, b2n = b2, b3n = b3;
        if (kblk + 1 < nk) {
            const __bf16* An = Arow + (size_t)(kblk + 1) * 32;
            __builtin_prefetch(An + 32, 0, 3);                  // global_prefetch_b8
            alo_n = *(const v8bf*)(An);
            ahi_n = *(const v8bf*)(An + 8);
            const __bf16* fbn = fbBase + (size_t)(kblk + 1) * fstep;
            b0n = *(const v16bf*)(fbn);
            b1n = *(const v16bf*)(fbn + 512);
            b2n = *(const v16bf*)(fbn + 1024);
            b3n = *(const v16bf*)(fbn + 1536);
        }
        // A fragment: element s = A(m, kblk*32 + 16*half + s), same k order as packed B
        v16bf av = __builtin_shufflevector(alo, ahi, 0, 1, 2, 3, 4, 5, 6, 7,
                                           8, 9, 10, 11, 12, 13, 14, 15);
        c0 = __builtin_amdgcn_wmma_f32_16x16x32_bf16(false, av, false, b0, (short)0, c0, false, false);
        c1 = __builtin_amdgcn_wmma_f32_16x16x32_bf16(false, av, false, b1, (short)0, c1, false, false);
        c2 = __builtin_amdgcn_wmma_f32_16x16x32_bf16(false, av, false, b2, (short)0, c2, false, false);
        c3 = __builtin_amdgcn_wmma_f32_16x16x32_bf16(false, av, false, b3, (short)0, c3, false, false);
        alo = alo_n; ahi = ahi_n;
        b0 = b0n; b1 = b1n; b2 = b2n; b3 = b3n;
    }
    // C/D layout: VGPR v -> row v (lanes 0-15) / row v+8 (lanes 16-31), col = l15
    #pragma unroll
    for (int j = 0; j < 4; ++j) {
        const v8f& cc = (j == 0) ? c0 : (j == 1) ? c1 : (j == 2) ? c2 : c3;
        int ncol = col0 + j * 16 + l15;
        float bb = bias ? bias[ncol] : 0.0f;
        #pragma unroll
        for (int v = 0; v < 8; ++v) {
            int m = row0 + v + half * 8;
            float r = cc[v] + bb;
            if (ACT == 1) r = fmaxf(r, 0.0f);
            C[(size_t)m * Nc + ncol] = r;
        }
    }
}

// ---------------- GAT: attention coefficients a_s / a_d ----------------
__global__ void gat_coef_kernel(const float* __restrict__ h, const float* __restrict__ a_src,
                                const float* __restrict__ a_dst,
                                float* __restrict__ as_, float* __restrict__ ad_)
{
    int idx = blockIdx.x * blockDim.x + threadIdx.x;   // (t*N+n)*4 + head
    if (idx >= NTN * HEADSC) return;
    int head = idx & 3;
    int nn   = idx >> 2;
    const float* hr = h + (size_t)nn * GOUT + head * HIDC;
    const float* asr = a_src + head * HIDC;
    const float* adr = a_dst + head * HIDC;
    float s = 0.f, d = 0.f;
    for (int f = 0; f < HIDC; ++f) { s += hr[f] * asr[f]; d += hr[f] * adr[f]; }
    as_[idx] = s;
    ad_[idx] = d;
}

__device__ __forceinline__ void edge_sd(const int* ei, int t, int e, int& src, int& dst) {
    if (e < EE) {
        src = ei[(size_t)t * 2 * EE + e];
        dst = ei[(size_t)t * 2 * EE + EE + e];
    } else {
        src = dst = e - EE;     // self loop
    }
}

__global__ void edge_max_kernel(const int* __restrict__ ei, const float* __restrict__ as_,
                                const float* __restrict__ ad_, unsigned* __restrict__ mx)
{
    int idx = blockIdx.x * blockDim.x + threadIdx.x;   // ((t*E2+e)*4 + h)
    if (idx >= TT * E2C * HEADSC) return;
    int h = idx & 3, rest = idx >> 2;
    int t = rest / E2C, e = rest % E2C;
    int src, dst; edge_sd(ei, t, e, src, dst);
    float a = as_[(size_t)(t * NND + src) * 4 + h] + ad_[(size_t)(t * NND + dst) * 4 + h];
    a = (a > 0.f) ? a : 0.2f * a;                      // leaky_relu 0.2
    atomicMax(&mx[(size_t)(t * NND + dst) * 4 + h], fenc(a));
}

__global__ void edge_exp_kernel(const int* __restrict__ ei, const float* __restrict__ as_,
                                const float* __restrict__ ad_, const unsigned* __restrict__ mx,
                                float* __restrict__ ebuf, float* __restrict__ dn)
{
    int idx = blockIdx.x * blockDim.x + threadIdx.x;
    if (idx >= TT * E2C * HEADSC) return;
    int h = idx & 3, rest = idx >> 2;
    int t = rest / E2C, e = rest % E2C;
    int src, dst; edge_sd(ei, t, e, src, dst);
    float a = as_[(size_t)(t * NND + src) * 4 + h] + ad_[(size_t)(t * NND + dst) * 4 + h];
    a = (a > 0.f) ? a : 0.2f * a;
    float m = fdec(mx[(size_t)(t * NND + dst) * 4 + h]);
    float ev = __expf(a - m);
    ebuf[idx] = ev;
    atomicAdd(&dn[(size_t)(t * NND + dst) * 4 + h], ev);
}

__global__ void edge_aggr_kernel(const int* __restrict__ ei, const float* __restrict__ hbuf,
                                 const float* __restrict__ ebuf, const float* __restrict__ dn,
                                 float* __restrict__ out)
{
    int eg = blockIdx.x;                    // t*E2 + e
    int t = eg / E2C, e = eg % E2C;
    int src, dst; edge_sd(ei, t, e, src, dst);
    int f = threadIdx.x;                    // 512 features
    int h = f >> 7;
    float w = ebuf[(size_t)eg * 4 + h] / dn[(size_t)(t * NND + dst) * 4 + h];
    atomicAdd(&out[(size_t)(t * NND + dst) * GOUT + f],
              hbuf[(size_t)(t * NND + src) * GOUT + f] * w);
}

__global__ void bias_relu_kernel(float* __restrict__ y, const float* __restrict__ bias,
                                 size_t total, int cols)
{
    size_t idx = (size_t)blockIdx.x * blockDim.x + threadIdx.x;
    if (idx >= total) return;
    y[idx] = fmaxf(y[idx] + bias[idx % cols], 0.0f);
}

// ---------------- GRU sequential scan (single workgroup) ----------------
// gi: [T*N, 384] precomputed (x@W_ih^T + b_ih). Output xout: [N, T, HID] row n*T+t.
__global__ void gru_scan_kernel(const float* __restrict__ gi, const float* __restrict__ Whh,
                                const float* __restrict__ bhh, float* __restrict__ xout)
{
    __shared__ float h[TT * HIDC];          // 1024
    __shared__ float gh[TT * 3 * HIDC];     // 3072
    int tid = threadIdx.x;                  // 256
    for (int i = tid; i < TT * HIDC; i += 256) h[i] = 0.0f;
    __syncthreads();
    for (int n = 0; n < NND; ++n) {
        for (int i = tid; i < TT * 3 * HIDC; i += 256) {
            int t = i / (3 * HIDC), j = i % (3 * HIDC);
            const float* w  = Whh + (size_t)j * HIDC;
            const float* hh = h + t * HIDC;
            float s = bhh[j];
            for (int f = 0; f < HIDC; ++f) s += hh[f] * w[f];
            gh[i] = s;
        }
        __syncthreads();
        for (int i = tid; i < TT * HIDC; i += 256) {
            int t = i / HIDC, f = i % HIDC;
            const float* gir = gi + (size_t)(t * NND + n) * (3 * HIDC);
            float ir = gir[f], iz = gir[HIDC + f], in_ = gir[2 * HIDC + f];
            float hr = gh[t * 3 * HIDC + f], hz = gh[t * 3 * HIDC + HIDC + f],
                  hn = gh[t * 3 * HIDC + 2 * HIDC + f];
            float r = sigmf(ir + hr);
            float z = sigmf(iz + hz);
            float nn = tanhf(in_ + r * hn);
            float hnew = (1.0f - z) * nn + z * h[i];
            h[i] = hnew;
            xout[(size_t)(n * TT + t) * HIDC + f] = hnew;
        }
        __syncthreads();
    }
}

// ---------------- transformer temporal attention (S=T=8, dh=32, 4 heads) ----------------
__global__ void tfm_attn_kernel(const float* __restrict__ qkv, float* __restrict__ attn)
{
    int idx = blockIdx.x * blockDim.x + threadIdx.x;   // n*NHTF + h
    if (idx >= NND * NHTF) return;
    int n = idx / NHTF, h = idx % NHTF;
    const int dh = HIDC / NHTF;                        // 32
    const float scale = rsqrtf((float)dh);
    float sc[TT][TT];
    for (int s = 0; s < TT; ++s) {
        const float* q = qkv + (size_t)(n * TT + s) * (3 * HIDC) + h * dh;
        for (int t2 = 0; t2 < TT; ++t2) {
            const float* k = qkv + (size_t)(n * TT + t2) * (3 * HIDC) + HIDC + h * dh;
            float a = 0.f;
            for (int d = 0; d < dh; ++d) a += q[d] * k[d];
            sc[s][t2] = a * scale;
        }
    }
    for (int s = 0; s < TT; ++s) {
        float m = sc[s][0];
        for (int t2 = 1; t2 < TT; ++t2) m = fmaxf(m, sc[s][t2]);
        float den = 0.f;
        for (int t2 = 0; t2 < TT; ++t2) { sc[s][t2] = __expf(sc[s][t2] - m); den += sc[s][t2]; }
        float inv = 1.0f / den;
        for (int t2 = 0; t2 < TT; ++t2) sc[s][t2] *= inv;
    }
    for (int s = 0; s < TT; ++s) {
        float* o = attn + (size_t)(n * TT + s) * HIDC + h * dh;
        for (int d = 0; d < dh; ++d) {
            float a = 0.f;
            for (int t2 = 0; t2 < TT; ++t2)
                a += sc[s][t2] * qkv[(size_t)(n * TT + t2) * (3 * HIDC) + 2 * HIDC + h * dh + d];
            o[d] = a;
        }
    }
}

// ---------------- residual + LayerNorm over 128 features (one wave per row) ----------------
__global__ void add_ln_kernel(float* __restrict__ x, const float* __restrict__ r,
                              const float* __restrict__ g, const float* __restrict__ b,
                              int rows)
{
    int wave = (blockIdx.x * blockDim.x + threadIdx.x) >> 5;
    int lane = threadIdx.x & 31;
    if (wave >= rows) return;
    float* xr = x + (size_t)wave * HIDC;
    const float* rr = r + (size_t)wave * HIDC;
    float v[4];
    float s = 0.f;
    #pragma unroll
    for (int i = 0; i < 4; ++i) {
        int col = lane + 32 * i;
        v[i] = xr[col] + rr[col];
        s += v[i];
    }
    float mu = wave_sum(s) * (1.0f / HIDC);
    float q = 0.f;
    #pragma unroll
    for (int i = 0; i < 4; ++i) { float d = v[i] - mu; q += d * d; }
    float var = wave_sum(q) * (1.0f / HIDC);
    float is = rsqrtf(var + LNEPS);
    #pragma unroll
    for (int i = 0; i < 4; ++i) {
        int col = lane + 32 * i;
        xr[col] = (v[i] - mu) * is * g[col] + b[col];
    }
}

// ---------------- SAGPool ----------------
__global__ void pool_score_init_kernel(const float* __restrict__ xtf,
                                       const float* __restrict__ Wrel,
                                       const float* __restrict__ Wroot,
                                       const float* __restrict__ brel,
                                       float* __restrict__ prel, float* __restrict__ score)
{
    int idx = blockIdx.x * blockDim.x + threadIdx.x;   // t*N + n
    if (idx >= TT * NND) return;
    int t = idx / NND, n = idx % NND;
    const float* xr = xtf + (size_t)(n * TT + t) * HIDC;
    float pr = 0.f, po = 0.f;
    for (int f = 0; f < HIDC; ++f) { pr += xr[f] * Wrel[f]; po += xr[f] * Wroot[f]; }
    prel[idx]  = pr;
    score[idx] = po + brel[0];
}

__global__ void pool_score_edges_kernel(const int* __restrict__ ei,
                                        const float* __restrict__ prel,
                                        float* __restrict__ score)
{
    int idx = blockIdx.x * blockDim.x + threadIdx.x;   // t*E + e (no self loops)
    if (idx >= TT * EE) return;
    int t = idx / EE, e = idx % EE;
    int src = ei[(size_t)t * 2 * EE + e];
    int dst = ei[(size_t)t * 2 * EE + EE + e];
    atomicAdd(&score[t * NND + dst], prel[t * NND + src]);
}

__global__ void topk_kernel(const float* __restrict__ score, int* __restrict__ perm,
                            float* __restrict__ vals)
{
    int t = blockIdx.x;
    __shared__ float sc[NND];
    for (int i = threadIdx.x; i < NND; i += blockDim.x) sc[i] = score[t * NND + i];
    __syncthreads();
    for (int n = threadIdx.x; n < NND; n += blockDim.x) {
        float v = sc[n];
        int rank = 0;
        for (int m = 0; m < NND; ++m)
            rank += (sc[m] > v) || (sc[m] == v && m < n);
        if (rank < KPOOL) {
            perm[t * KPOOL + rank] = n;
            vals[t * KPOOL + rank] = v;
        }
    }
}

__global__ void pool_gather_kernel(const float* __restrict__ xtf, const int* __restrict__ perm,
                                   const float* __restrict__ vals, float* __restrict__ pooled)
{
    int idx = blockIdx.x * blockDim.x + threadIdx.x;   // t*K*H + k*H + f
    if (idx >= TT * KPOOL * HIDC) return;
    int t = idx / (KPOOL * HIDC);
    int r = idx % (KPOOL * HIDC);
    int k = r / HIDC, f = r % HIDC;
    int n = perm[t * KPOOL + k];
    pooled[idx] = xtf[(size_t)(n * TT + t) * HIDC + f] * tanhf(vals[t * KPOOL + k]);
}

// ---------------- final temporal MHA (nh=1, dh=128, S=T=8, B=KPOOL) ----------------
__global__ void final_attn_kernel(const float* __restrict__ qkv2, float* __restrict__ attn2)
{
    int b = blockIdx.x * blockDim.x + threadIdx.x;
    if (b >= KPOOL) return;
    const float scale = rsqrtf((float)HIDC);
    float sc[TT][TT];
    for (int s = 0; s < TT; ++s) {
        const float* q = qkv2 + (size_t)(s * KPOOL + b) * (3 * HIDC);
        for (int t2 = 0; t2 < TT; ++t2) {
            const float* k = qkv2 + (size_t)(t2 * KPOOL + b) * (3 * HIDC) + HIDC;
            float a = 0.f;
            for (int d = 0; d < HIDC; ++d) a += q[d] * k[d];
            sc[s][t2] = a * scale;
        }
    }
    for (int s = 0; s < TT; ++s) {
        float m = sc[s][0];
        for (int t2 = 1; t2 < TT; ++t2) m = fmaxf(m, sc[s][t2]);
        float den = 0.f;
        for (int t2 = 0; t2 < TT; ++t2) { sc[s][t2] = __expf(sc[s][t2] - m); den += sc[s][t2]; }
        float inv = 1.0f / den;
        for (int t2 = 0; t2 < TT; ++t2) sc[s][t2] *= inv;
    }
    for (int s = 0; s < TT; ++s) {
        float* o = attn2 + (size_t)(s * KPOOL + b) * HIDC;
        for (int d = 0; d < HIDC; ++d) {
            float a = 0.f;
            for (int t2 = 0; t2 < TT; ++t2)
                a += sc[s][t2] * qkv2[(size_t)(t2 * KPOOL + b) * (3 * HIDC) + 2 * HIDC + d];
            o[d] = a;
        }
    }
}

__global__ void mean_kernel(const float* __restrict__ proj2, float* __restrict__ out)
{
    int idx = blockIdx.x * blockDim.x + threadIdx.x;   // k*HID + f
    if (idx >= KPOOL * HIDC) return;
    int k = idx / HIDC, f = idx % HIDC;
    float s = 0.f;
    for (int t = 0; t < TT; ++t) s += proj2[(size_t)(t * KPOOL + k) * HIDC + f];
    out[idx] = s * (1.0f / TT);
}

// ---------------- launch ----------------
extern "C" void kernel_launch(void* const* d_in, const int* in_sizes, int n_in,
                              void* d_out, int out_size, void* d_ws, size_t ws_size,
                              hipStream_t stream)
{
    (void)in_sizes; (void)n_in; (void)out_size; (void)ws_size;
    // ---- inputs, setup_inputs() dict order ----
    const float* feats = (const float*)d_in[0];        // [T,N,64]
    const int*   ei    = (const int*)d_in[1];          // [T,2,E]
    const float* gatW[2]  = { (const float*)d_in[2], (const float*)d_in[6] };
    const float* gatAs[2] = { (const float*)d_in[3], (const float*)d_in[7] };
    const float* gatAd[2] = { (const float*)d_in[4], (const float*)d_in[8] };
    const float* gatB[2]  = { (const float*)d_in[5], (const float*)d_in[9] };
    const float* Wih = (const float*)d_in[10];         // [384,512]
    const float* Whh = (const float*)d_in[11];         // [384,128]
    const float* bih = (const float*)d_in[12];
    const float* bhh = (const float*)d_in[13];
    // tfm layers: base 14 + 12*layer
    const float* Wrel  = (const float*)d_in[38];
    const float* brel  = (const float*)d_in[39];
    const float* Wroot = (const float*)d_in[40];
    const float* mWqkv = (const float*)d_in[41];
    const float* mbqkv = (const float*)d_in[42];
    const float* mWo   = (const float*)d_in[43];
    const float* mbo   = (const float*)d_in[44];
    float* out = (float*)d_out;

    // ---- workspace arena (float offsets) ----
    float* W = (float*)d_ws;
    const size_t SZH = (size_t)NTN * GOUT;             // 8,388,608
    float*    buf0    = W;                             // GAT h
    float*    buf1    = W + SZH;                       // GAT layer-0 out / layer-1 in
    float*    spatial = W + 2 * SZH;                   // GAT layer-1 out
    float*    as_     = W + 3 * SZH;                   // NTN*4
    float*    ad_     = as_ + (size_t)NTN * 4;
    unsigned* mx      = (unsigned*)(ad_ + (size_t)NTN * 4);
    float*    dn      = (float*)mx + (size_t)NTN * 4;
    float*    ebuf    = dn + (size_t)NTN * 4;          // T*E2*4 = 1,114,112
    float*    hiwater = ebuf + (size_t)TT * E2C * 4;   // ~26.5M floats
    // bf16 staging (beyond stage-1 high water): A (max NTN*512) + packed B (max 512*512)
    __bf16* Abf = (__bf16*)hiwater;                    // 8,388,608 bf16 = 4.2M floats
    __bf16* Bpk = (__bf16*)(hiwater + SZH / 2);        // 262,144 bf16
    // stage 2: reuse [0, 2*SZH)
    float* gi    = W;                                  // NTN*384
    float* xtf   = W + (size_t)NTN * 384;              // N*T*128 = 2,097,152
    float* qkv   = xtf + (size_t)NTN * HIDC;           // NTN*384
    float* attnb = qkv + (size_t)NTN * 384;            // NTN*128
    float* tmp   = attnb + (size_t)NTN * HIDC;         // NTN*128 (spatial region, post-GRU)
    float* ffn1  = tmp + (size_t)NTN * HIDC;           // NTN*256
    float* prel  = ffn1 + (size_t)NTN * 256;           // T*N
    float* score = prel + (size_t)TT * NND;
    float* vals  = score + (size_t)TT * NND;           // T*K
    float* pooled= vals + (size_t)TT * KPOOL;          // T*K*128
    float* qkv2  = pooled + (size_t)TT * KPOOL * HIDC; // 2048*384
    float* attn2 = qkv2 + (size_t)2048 * 384;          // 2048*128
    float* proj2 = attn2 + (size_t)2048 * HIDC;        // 2048*128
    int*   perm  = (int*)(proj2 + (size_t)2048 * HIDC);// T*K ints

    // C = act(A @ opB + bias): stage A->bf16, pack B->fragments, WMMA GEMM.
    auto gemm = [&](const float* A, const float* B, const float* bias, float* C,
                    int M, int Nc, int K, int ldb, int transB, int act) {
        int n4 = (M * K) / 4;
        cvt_bf16_kernel<<<(n4 + 255) / 256, 256, 0, stream>>>(A, Abf, n4);
        int pthreads = (K / 32) * (Nc / 16) * 32;
        pack_b_kernel<<<(pthreads + 255) / 256, 256, 0, stream>>>(B, Bpk, Nc, K, ldb, transB);
        int strips = (M / 16) * (Nc / 64);             // 16x64 strip per wave
        int grid   = (strips + 7) / 8;
        if (act)
            wmma_gemm_kernel<1><<<grid, 256, 0, stream>>>(Abf, Bpk, bias, C, M, Nc, K);
        else
            wmma_gemm_kernel<0><<<grid, 256, 0, stream>>>(Abf, Bpk, bias, C, M, Nc, K);
    };

    // ================= GAT layers =================
    const float* layerIn[2] = { feats, buf1 };
    const int layerK[2] = { INDIM, GOUT };
    float* layerOut[2] = { buf1, spatial };
    const int nEdgeThreads = TT * E2C * HEADSC;
    for (int l = 0; l < 2; ++l) {
        // h = X @ W   -> buf0
        gemm(layerIn[l], gatW[l], nullptr, buf0, NTN, GOUT, layerK[l], GOUT, 0, 0);
        // attention coefficients
        gat_coef_kernel<<<(NTN * HEADSC + 255) / 256, 256, 0, stream>>>(buf0, gatAs[l], gatAd[l], as_, ad_);
        // edge softmax
        hipMemsetAsync(mx, 0, (size_t)NTN * 4 * sizeof(unsigned), stream);
        hipMemsetAsync(dn, 0, (size_t)NTN * 4 * sizeof(float), stream);
        edge_max_kernel<<<(nEdgeThreads + 255) / 256, 256, 0, stream>>>(ei, as_, ad_, mx);
        edge_exp_kernel<<<(nEdgeThreads + 255) / 256, 256, 0, stream>>>(ei, as_, ad_, mx, ebuf, dn);
        // weighted aggregation
        hipMemsetAsync(layerOut[l], 0, SZH * sizeof(float), stream);
        edge_aggr_kernel<<<TT * E2C, GOUT, 0, stream>>>(ei, buf0, ebuf, dn, layerOut[l]);
        bias_relu_kernel<<<(int)((SZH + 255) / 256), 256, 0, stream>>>(layerOut[l], gatB[l], SZH, GOUT);
    }

    // ================= GRU =================
    gemm(spatial, Wih, bih, gi, NTN, 3 * HIDC, GOUT, GOUT, 1, 0);   // gi = x @ W_ih^T + b_ih
    gru_scan_kernel<<<1, 256, 0, stream>>>(gi, Whh, bhh, xtf);      // xtf: [N,T,128]

    // ================= Transformer encoder (2 layers, post-LN) =================
    for (int l = 0; l < 2; ++l) {
        const float** p = (const float**)&d_in[14 + 12 * l];
        const float *Wqkv = p[0], *bqkv = p[1], *Wo = p[2], *bo = p[3];
        const float *ln1g = p[4], *ln1b = p[5], *W1 = p[6], *b1 = p[7];
        const float *W2 = p[8], *b2 = p[9], *ln2g = p[10], *ln2b = p[11];
        gemm(xtf, Wqkv, bqkv, qkv, NTN, 3 * HIDC, HIDC, HIDC, 1, 0);
        tfm_attn_kernel<<<(NND * NHTF + 255) / 256, 256, 0, stream>>>(qkv, attnb);
        gemm(attnb, Wo, bo, tmp, NTN, HIDC, HIDC, HIDC, 1, 0);
        add_ln_kernel<<<(NTN * 32 + 255) / 256, 256, 0, stream>>>(xtf, tmp, ln1g, ln1b, NTN);
        gemm(xtf, W1, b1, ffn1, NTN, 2 * HIDC, HIDC, HIDC, 1, 1);    // relu
        gemm(ffn1, W2, b2, tmp, NTN, HIDC, 2 * HIDC, 2 * HIDC, 1, 0);
        add_ln_kernel<<<(NTN * 32 + 255) / 256, 256, 0, stream>>>(xtf, tmp, ln2g, ln2b, NTN);
    }

    // ================= SAGPool per timestep =================
    pool_score_init_kernel<<<(TT * NND + 255) / 256, 256, 0, stream>>>(xtf, Wrel, Wroot, brel, prel, score);
    pool_score_edges_kernel<<<(TT * EE + 255) / 256, 256, 0, stream>>>(ei, prel, score);
    topk_kernel<<<TT, 256, 0, stream>>>(score, perm, vals);
    pool_gather_kernel<<<(TT * KPOOL * HIDC + 255) / 256, 256, 0, stream>>>(xtf, perm, vals, pooled);

    // ================= final temporal MHA + mean =================
    gemm(pooled, mWqkv, mbqkv, qkv2, TT * KPOOL, 3 * HIDC, HIDC, HIDC, 1, 0);
    final_attn_kernel<<<(KPOOL + 255) / 256, 256, 0, stream>>>(qkv2, attn2);
    gemm(attn2, mWo, mbo, proj2, TT * KPOOL, HIDC, HIDC, HIDC, 1, 0);
    mean_kernel<<<(KPOOL * HIDC + 255) / 256, 256, 0, stream>>>(proj2, out);
}
